// GAT_11785390260547
// MI455X (gfx1250) — matrix operand order repaired and checked by reading, not compiled, not run on hardware
//
#include <hip/hip_runtime.h>
#include <hip/hip_bf16.h>
#include <math.h>

#define N_NODES 50000
#define N_EDGES 1000000
#define IN_F    128
#define C1F     32
#define HEADS   4
#define F1      128   // HEADS * C1F
#define C2F     64
#define OUT_F   8
#define BN_EPS  1e-5f
#define NSLOPE  0.2f

typedef __attribute__((ext_vector_type(16))) _Float16 v16h;
typedef __attribute__((ext_vector_type(8)))  _Float16 v8h;
typedef __attribute__((ext_vector_type(8)))  float    v8f;

// ---- monotonic float<->uint for atomicMax-based segment max ----
__device__ __forceinline__ unsigned ford(float f) {
  unsigned u = __float_as_uint(f);
  return (u & 0x80000000u) ? ~u : (u | 0x80000000u);
}
__device__ __forceinline__ float finv(unsigned u) {
  return __uint_as_float((u & 0x80000000u) ? (u & 0x7FFFFFFFu) : ~u);
}

// ---- mean(edge_attr) reduction ----
__global__ void k_easum(const float* __restrict__ ea, float* __restrict__ easum, int n) {
  float s = 0.f;
  for (int i = blockIdx.x * blockDim.x + threadIdx.x; i < n; i += gridDim.x * blockDim.x)
    s += ea[i];
  #pragma unroll
  for (int off = 16; off; off >>= 1) s += __shfl_xor(s, off, 32);
  if ((threadIdx.x & 31) == 0) atomicAdd(easum, s);
}

// ---- fp32 -> fp16 elementwise ----
__global__ void k_cvt_f16(const float* __restrict__ in, _Float16* __restrict__ out, int n) {
  int i = blockIdx.x * blockDim.x + threadIdx.x;
  if (i < n) out[i] = (_Float16)in[i];
}

// ---- W[K x Nout] fp32 -> WT[Nout x K] fp16 ----
__global__ void k_wT_f16(const float* __restrict__ W, _Float16* __restrict__ WT, int K, int Nout) {
  int i = blockIdx.x * blockDim.x + threadIdx.x;
  if (i >= K * Nout) return;
  int k = i / Nout, n = i - k * Nout;
  WT[(size_t)n * K + k] = (_Float16)W[i];
}

// ---- WMMA GEMM: C[MxNout] = A[MxK](f16) @ BT[NoutxK]^T(f16) + bias, f32 accumulate ----
// one wave per 16x16 output tile; K stepped by 32 via v_wmma_f32_16x16x32_f16
__global__ void k_gemm_wmma(const _Float16* __restrict__ A, const _Float16* __restrict__ BT,
                            const float* __restrict__ bias, float* __restrict__ C,
                            int M, int K, int Nout) {
  int lane = threadIdx.x & 31;
  int wave = (int)((blockIdx.x * blockDim.x + threadIdx.x) >> 5);
  int ntiles = Nout >> 4;
  int mtiles = M >> 4;
  if (wave >= mtiles * ntiles) return;
  int mt = wave / ntiles, nt = wave - mt * ntiles;

  int arow = mt * 16 + (lane & 15);
  int bcol = nt * 16 + (lane & 15);
  const _Float16* ap = A  + (size_t)arow * K + ((lane >> 4) * 16);  // A frag: 16 contiguous K
  const _Float16* bp = BT + (size_t)bcol * K + ((lane >> 4) * 16);  // B frag: 16 contiguous K

  v8f acc = {};
  for (int k0 = 0; k0 < K; k0 += 32) {
    if (k0 + 32 < K) {
      __builtin_prefetch(ap + k0 + 32, 0, 0);
      __builtin_prefetch(bp + k0 + 32, 0, 0);
    }
    v8h a0 = *(const v8h*)(ap + k0);
    v8h a1 = *(const v8h*)(ap + k0 + 8);
    v16h a, b;
    b = *(const v16h*)(bp + k0);
    #pragma unroll
    for (int i = 0; i < 8; i++) { a[i] = a0[i]; a[8 + i] = a1[i]; }
    acc = __builtin_amdgcn_wmma_f32_16x16x32_f16(false, a, false, b, (short)0, acc, false, false);
  }

  int crow = mt * 16 + ((lane >> 4) << 3);   // VGPR v -> row crow+v
  float bv = bias[bcol];
  #pragma unroll
  for (int v = 0; v < 8; v++)
    C[(size_t)(crow + v) * Nout + bcol] = acc[v] + bv;
}

// ---- layer1 attention logits: wave per edge, lane = channel within head ----
__global__ void k_att1(const int* __restrict__ src, const int* __restrict__ dst,
                       const float* __restrict__ ea, const float* __restrict__ easum,
                       const float* __restrict__ xl, const float* __restrict__ xr,
                       const float* __restrict__ We, const float* __restrict__ att,
                       float* __restrict__ alog, unsigned* __restrict__ amax) {
  int lane = threadIdx.x & 31;
  int e = (int)((blockIdx.x * blockDim.x + threadIdx.x) >> 5);
  const int TOT = N_EDGES + N_NODES;
  if (e >= TOT) return;
  int s = (e < N_EDGES) ? src[e] : (e - N_EDGES);
  int d = (e < N_EDGES) ? dst[e] : (e - N_EDGES);
  float eav = (e < N_EDGES) ? ea[e] : (easum[0] * (1.0f / N_EDGES));
  const float* xls = xl + (size_t)s * F1;
  const float* xrd = xr + (size_t)d * F1;
  #pragma unroll
  for (int h = 0; h < HEADS; h++) {
    int f = h * C1F + lane;
    float m = xls[f] + xrd[f] + eav * We[f];
    m = (m > 0.f) ? m : NSLOPE * m;
    float v = m * att[f];
    #pragma unroll
    for (int off = 16; off; off >>= 1) v += __shfl_xor(v, off, 32);
    if (lane == 0) {
      alog[(size_t)e * HEADS + h] = v;
      atomicMax(&amax[d * HEADS + h], ford(v));
    }
  }
}

// ---- generic scatter-softmax: exp(a - max) and denominator accumulate ----
__global__ void k_softmax_edge(const int* __restrict__ dst, const unsigned* __restrict__ amax,
                               float* __restrict__ alog, float* __restrict__ den,
                               int Hn) {
  int i = blockIdx.x * blockDim.x + threadIdx.x;
  int tot = (N_EDGES + N_NODES) * Hn;
  if (i >= tot) return;
  int e = i / Hn, h = i - e * Hn;
  int d = (e < N_EDGES) ? dst[e] : (e - N_EDGES);
  float ex = __expf(alog[i] - finv(amax[d * Hn + h]));
  alog[i] = ex;
  atomicAdd(&den[d * Hn + h], ex);
}

// ---- layer1 weighted scatter-aggregate ----
__global__ void k_agg1(const int* __restrict__ src, const int* __restrict__ dst,
                       const float* __restrict__ xl, const float* __restrict__ alog,
                       const float* __restrict__ den, float* __restrict__ out) {
  int lane = threadIdx.x & 31;
  int e = (int)((blockIdx.x * blockDim.x + threadIdx.x) >> 5);
  const int TOT = N_EDGES + N_NODES;
  if (e >= TOT) return;
  int s = (e < N_EDGES) ? src[e] : (e - N_EDGES);
  int d = (e < N_EDGES) ? dst[e] : (e - N_EDGES);
  #pragma unroll
  for (int h = 0; h < HEADS; h++) {
    float alpha = alog[(size_t)e * HEADS + h] / den[d * HEADS + h];
    int f = h * C1F + lane;
    atomicAdd(&out[(size_t)d * F1 + f], xl[(size_t)s * F1 + f] * alpha);
  }
}

// ---- layer2 (H=1, C=64) attention logits ----
__global__ void k_att2(const int* __restrict__ src, const int* __restrict__ dst,
                       const float* __restrict__ ea, const float* __restrict__ easum,
                       const float* __restrict__ xl, const float* __restrict__ xr,
                       const float* __restrict__ We, const float* __restrict__ att,
                       float* __restrict__ alog, unsigned* __restrict__ amax) {
  int lane = threadIdx.x & 31;
  int e = (int)((blockIdx.x * blockDim.x + threadIdx.x) >> 5);
  const int TOT = N_EDGES + N_NODES;
  if (e >= TOT) return;
  int s = (e < N_EDGES) ? src[e] : (e - N_EDGES);
  int d = (e < N_EDGES) ? dst[e] : (e - N_EDGES);
  float eav = (e < N_EDGES) ? ea[e] : (easum[0] * (1.0f / N_EDGES));
  const float* xls = xl + (size_t)s * C2F;
  const float* xrd = xr + (size_t)d * C2F;
  float m0 = xls[lane] + xrd[lane] + eav * We[lane];
  m0 = (m0 > 0.f) ? m0 : NSLOPE * m0;
  float m1 = xls[lane + 32] + xrd[lane + 32] + eav * We[lane + 32];
  m1 = (m1 > 0.f) ? m1 : NSLOPE * m1;
  float v = m0 * att[lane] + m1 * att[lane + 32];
  #pragma unroll
  for (int off = 16; off; off >>= 1) v += __shfl_xor(v, off, 32);
  if (lane == 0) {
    alog[e] = v;
    atomicMax(&amax[d], ford(v));
  }
}

// ---- layer2 weighted scatter-aggregate ----
__global__ void k_agg2(const int* __restrict__ src, const int* __restrict__ dst,
                       const float* __restrict__ xl, const float* __restrict__ alog,
                       const float* __restrict__ den, float* __restrict__ out) {
  int lane = threadIdx.x & 31;
  int e = (int)((blockIdx.x * blockDim.x + threadIdx.x) >> 5);
  const int TOT = N_EDGES + N_NODES;
  if (e >= TOT) return;
  int s = (e < N_EDGES) ? src[e] : (e - N_EDGES);
  int d = (e < N_EDGES) ? dst[e] : (e - N_EDGES);
  float alpha = alog[e] / den[d];
  atomicAdd(&out[(size_t)d * C2F + lane],      xl[(size_t)s * C2F + lane]      * alpha);
  atomicAdd(&out[(size_t)d * C2F + lane + 32], xl[(size_t)s * C2F + lane + 32] * alpha);
}

// ---- relu(x + bias) in place + per-column sum / sumsq (block-partial then atomic) ----
__global__ void k_relu_bias_stats(float* __restrict__ y, const float* __restrict__ bias,
                                  float* __restrict__ sums, int n, int F) {
  int f = threadIdx.x;  // blockDim.x == F
  int rows = (n + gridDim.x - 1) / gridDim.x;
  int r0 = blockIdx.x * rows;
  int r1 = r0 + rows; if (r1 > n) r1 = n;
  float b = bias[f], s = 0.f, ss = 0.f;
  for (int r = r0; r < r1; r++) {
    size_t idx = (size_t)r * F + f;
    float v = y[idx] + b;
    v = (v > 0.f) ? v : 0.f;
    y[idx] = v;
    s += v; ss += v * v;
  }
  atomicAdd(&sums[f], s);
  atomicAdd(&sums[F + f], ss);
}

// ---- BatchNorm apply; optional fp32 and/or fp16 outputs ----
__global__ void k_bn_apply(const float* __restrict__ y, const float* __restrict__ sums,
                           const float* __restrict__ g, const float* __restrict__ bt,
                           float* __restrict__ out, _Float16* __restrict__ out16,
                           int n, int F) {
  int i = blockIdx.x * blockDim.x + threadIdx.x;
  if (i >= n * F) return;
  int f = i % F;
  float inv_n = 1.0f / (float)n;
  float mean = sums[f] * inv_n;
  float var  = sums[F + f] * inv_n - mean * mean;
  float sc = g[f] * rsqrtf(var + BN_EPS);
  float v = (y[i] - mean) * sc + bt[f];
  if (out)   out[i]   = v;
  if (out16) out16[i] = (_Float16)v;
}

// ---- final linear 64 -> 8 ----
__global__ void k_linear(const float* __restrict__ h, const float* __restrict__ W,
                         const float* __restrict__ b, float* __restrict__ out) {
  int i = blockIdx.x * blockDim.x + threadIdx.x;
  if (i >= N_NODES * OUT_F) return;
  int n = i / OUT_F, o = i - n * OUT_F;
  float s = b[o];
  #pragma unroll
  for (int k = 0; k < C2F; k++) s += h[(size_t)n * C2F + k] * W[k * OUT_F + o];
  out[i] = s;
}

extern "C" void kernel_launch(void* const* d_in, const int* in_sizes, int n_in,
                              void* d_out, int out_size, void* d_ws, size_t ws_size,
                              hipStream_t stream) {
  const float* x    = (const float*)d_in[0];
  const int*   ei   = (const int*)  d_in[1];
  const float* ea   = (const float*)d_in[2];
  const float* W1l  = (const float*)d_in[3];
  const float* b1l  = (const float*)d_in[4];
  const float* W1r  = (const float*)d_in[5];
  const float* b1r  = (const float*)d_in[6];
  const float* We1  = (const float*)d_in[7];
  const float* att1 = (const float*)d_in[8];
  const float* bias1= (const float*)d_in[9];
  const float* g1   = (const float*)d_in[10];
  const float* bt1  = (const float*)d_in[11];
  const float* W2l  = (const float*)d_in[12];
  const float* b2l  = (const float*)d_in[13];
  const float* W2r  = (const float*)d_in[14];
  const float* b2r  = (const float*)d_in[15];
  const float* We2  = (const float*)d_in[16];
  const float* att2 = (const float*)d_in[17];
  const float* bias2= (const float*)d_in[18];
  const float* g2   = (const float*)d_in[19];
  const float* bt2  = (const float*)d_in[20];
  const float* Wlin = (const float*)d_in[21];
  const float* blin = (const float*)d_in[22];
  const int* src = ei;
  const int* dst = ei + N_EDGES;

  // workspace carve-up (reused across layers)
  char* base = (char*)d_ws;
  size_t off = 0;
  auto take = [&](size_t bytes) -> void* {
    void* p = base + off;
    off = (off + bytes + 255) & ~(size_t)255;
    return p;
  };
  _Float16* a16  = (_Float16*)take((size_t)N_NODES * F1 * 2);           // x16 / h1bn16
  _Float16* wt16 = (_Float16*)take((size_t)128 * 128 * 2);              // transposed weights (f16)
  float*    xl   = (float*)   take((size_t)N_NODES * F1 * 4);           // left transform (both layers)
  float*    xr   = (float*)   take((size_t)N_NODES * F1 * 4);           // right transform
  float*    alog = (float*)   take((size_t)(N_EDGES + N_NODES) * HEADS * 4);
  unsigned* amax = (unsigned*)take((size_t)N_NODES * HEADS * 4);
  float*    den  = (float*)   take((size_t)N_NODES * HEADS * 4);
  float*    acc  = (float*)   take((size_t)N_NODES * F1 * 4);           // scatter accumulator
  float*    hbn  = (float*)   take((size_t)N_NODES * C2F * 4);          // layer-2 BN output
  float*    sums = (float*)   take((size_t)2 * F1 * 4);
  float*    easum= (float*)   take((size_t)256);

  const int EWAVES  = N_EDGES + N_NODES;          // 1,050,000
  const int EBLOCKS = (EWAVES + 7) / 8;           // 8 waves / 256-thread block

  // ---------------- layer 1 ----------------
  hipMemsetAsync(easum, 0, 4, stream);
  hipMemsetAsync(amax,  0, (size_t)N_NODES * HEADS * 4, stream);
  hipMemsetAsync(den,   0, (size_t)N_NODES * HEADS * 4, stream);
  hipMemsetAsync(acc,   0, (size_t)N_NODES * F1 * 4, stream);
  hipMemsetAsync(sums,  0, (size_t)2 * F1 * 4, stream);

  k_easum<<<1024, 256, 0, stream>>>(ea, easum, N_EDGES);
  k_cvt_f16<<<(N_NODES * IN_F + 255) / 256, 256, 0, stream>>>(x, a16, N_NODES * IN_F);

  int tiles1 = (N_NODES / 16) * (F1 / 16);
  k_wT_f16<<<(IN_F * F1 + 255) / 256, 256, 0, stream>>>(W1l, wt16, IN_F, F1);
  k_gemm_wmma<<<(tiles1 + 7) / 8, 256, 0, stream>>>(a16, wt16, b1l, xl, N_NODES, IN_F, F1);
  k_wT_f16<<<(IN_F * F1 + 255) / 256, 256, 0, stream>>>(W1r, wt16, IN_F, F1);
  k_gemm_wmma<<<(tiles1 + 7) / 8, 256, 0, stream>>>(a16, wt16, b1r, xr, N_NODES, IN_F, F1);

  k_att1<<<EBLOCKS, 256, 0, stream>>>(src, dst, ea, easum, xl, xr, We1, att1, alog, amax);
  k_softmax_edge<<<(EWAVES * HEADS + 255) / 256, 256, 0, stream>>>(dst, amax, alog, den, HEADS);
  k_agg1<<<EBLOCKS, 256, 0, stream>>>(src, dst, xl, alog, den, acc);

  k_relu_bias_stats<<<512, F1, 0, stream>>>(acc, bias1, sums, N_NODES, F1);
  k_bn_apply<<<(N_NODES * F1 + 255) / 256, 256, 0, stream>>>(acc, sums, g1, bt1,
                                                             nullptr, a16, N_NODES, F1);

  // ---------------- layer 2 ----------------
  hipMemsetAsync(amax, 0, (size_t)N_NODES * 4, stream);
  hipMemsetAsync(den,  0, (size_t)N_NODES * 4, stream);
  hipMemsetAsync(acc,  0, (size_t)N_NODES * C2F * 4, stream);
  hipMemsetAsync(sums, 0, (size_t)2 * C2F * 4, stream);

  int tiles2 = (N_NODES / 16) * (C2F / 16);
  k_wT_f16<<<(F1 * C2F + 255) / 256, 256, 0, stream>>>(W2l, wt16, F1, C2F);
  k_gemm_wmma<<<(tiles2 + 7) / 8, 256, 0, stream>>>(a16, wt16, b2l, xl, N_NODES, F1, C2F);
  k_wT_f16<<<(F1 * C2F + 255) / 256, 256, 0, stream>>>(W2r, wt16, F1, C2F);
  k_gemm_wmma<<<(tiles2 + 7) / 8, 256, 0, stream>>>(a16, wt16, b2r, xr, N_NODES, F1, C2F);

  k_att2<<<EBLOCKS, 256, 0, stream>>>(src, dst, ea, easum, xl, xr, We2, att2, alog, amax);
  k_softmax_edge<<<(EWAVES + 255) / 256, 256, 0, stream>>>(dst, amax, alog, den, 1);
  k_agg2<<<EBLOCKS, 256, 0, stream>>>(src, dst, xl, alog, den, acc);

  k_relu_bias_stats<<<512, C2F, 0, stream>>>(acc, bias2, sums, N_NODES, C2F);
  k_bn_apply<<<(N_NODES * C2F + 255) / 256, 256, 0, stream>>>(acc, sums, g2, bt2,
                                                              hbn, nullptr, N_NODES, C2F);

  k_linear<<<(N_NODES * OUT_F + 255) / 256, 256, 0, stream>>>(hbn, Wlin, blin, (float*)d_out);
}
